// AttentionModel_39848706572798
// MI455X (gfx1250) — matrix operand order repaired and checked by reading, not compiled, not run on hardware
//
#include <hip/hip_runtime.h>
#include <hip/hip_bf16.h>

// ---------------------------------------------------------------------------
// Attention (B=4, S=4096, E=1024) for MI455X / gfx1250, wave32 + WMMA bf16.
//   Pass 0: qkv_gemm x3 : Y = X @ W^T + b  (f32 in -> bf16 ws; V stored [B][E][S])
//   Pass 1: attn_stats  : per-row softmax max / sumexp (M, L)
//   Pass 2: attn_out    : recompute scores per 256-wide embed slice, P@V
// All contractions via v_wmma_f32_16x16x32_bf16; tile fills use async
// global->LDS (ASYNCcnt) double-buffered in 64-wide K chunks; all fragment
// reads are 2 x ds_load_b128 (16B-aligned, conflict-free LDS strides).
// ---------------------------------------------------------------------------

#define EMBED   1024
#define SEQ     4096
#define NBATCH  4
#define SCALE   0.03125f      // 1/sqrt(1024)
#define ETILE   256           // embed slice per attn_out block
#define KCHUNK  64            // staged K extent per double-buffer step
#define NCHUNK  (EMBED / KCHUNK)

typedef __attribute__((ext_vector_type(16))) __bf16 v16bf;
typedef __attribute__((ext_vector_type(8)))  __bf16 v8bf;
typedef __attribute__((ext_vector_type(8)))  float  v8f;
typedef __attribute__((__vector_size__(4 * sizeof(int)))) int v4i;

// ---- async global->LDS (gfx1250) with synchronous fallback ----------------
#if defined(__has_builtin)
#if __has_builtin(__builtin_amdgcn_global_load_async_to_lds_b128)
#define USE_ASYNC 1
#endif
#endif
#ifndef USE_ASYNC
#define USE_ASYNC 0
#endif

#if USE_ASYNC
typedef __attribute__((address_space(1))) v4i gas_v4i;
typedef __attribute__((address_space(3))) v4i las_v4i;
#endif

__device__ inline void copy16(const __bf16* g, __bf16* l) {
#if USE_ASYNC
  __builtin_amdgcn_global_load_async_to_lds_b128(
      (gas_v4i*)(void*)g, (las_v4i*)(void*)l, 0, 0);
#else
  *(v8bf*)l = *(const v8bf*)g;
#endif
}

template <int N>
__device__ inline void async_wait() {
#if USE_ASYNC
#if defined(__has_builtin) && __has_builtin(__builtin_amdgcn_s_wait_asynccnt)
  __builtin_amdgcn_s_wait_asynccnt(N);
#else
  asm volatile("s_wait_asynccnt %0" ::"i"(N) : "memory");
#endif
#endif
}

// ---- fragment builders (ISA 7.12.2 layouts, wave32, 2 x b128 each) --------

// A 16x32 bf16 from row-major [m][k] LDS; ld*2B and base must be 16B-aligned.
__device__ inline v16bf load_a_frag(const __bf16* base, int ld, int lane) {
  const int half = lane >> 4, ml = lane & 15;
  const __bf16* rp = base + ml * ld + 8 * half;
  const v8bf lo = *(const v8bf*)(rp);       // K 8*half + 0..7
  const v8bf hi = *(const v8bf*)(rp + 16);  // K 8*half + 16..23
  return __builtin_shufflevector(lo, hi, 0, 1, 2, 3, 4, 5, 6, 7,
                                 8, 9, 10, 11, 12, 13, 14, 15);
}

// B 32x16 bf16 from [n][k] LDS storage (N-major).
__device__ inline v16bf load_b_frag(const __bf16* base, int ld, int lane) {
  const int nl = lane & 15, kb = (lane >> 4) * 8;
  const __bf16* rp = base + nl * ld + kb;
  const v8bf lo = *(const v8bf*)(rp);
  const v8bf hi = *(const v8bf*)(rp + 16);
  return __builtin_shufflevector(lo, hi, 0, 1, 2, 3, 4, 5, 6, 7,
                                 8, 9, 10, 11, 12, 13, 14, 15);
}

// Q/K 64x64 double-buffered chunk fill: 8 x 16B async copies per thread.
__device__ inline void fill_qk(const __bf16* Qp, const __bf16* Kj, int tid,
                               __bf16 (*Qc)[72], __bf16 (*Kc)[72], int ec) {
  const int row = tid >> 1, koff = (tid & 1) * 32;
  const __bf16* qs = Qp + (size_t)row * EMBED + ec + koff;
  const __bf16* ks = Kj + (size_t)row * EMBED + ec + koff;
#pragma unroll
  for (int c = 0; c < 4; ++c) {
    copy16(qs + c * 8, &Qc[row][koff + c * 8]);
    copy16(ks + c * 8, &Kc[row][koff + c * 8]);
  }
}

// ---- Pass 0: projection GEMM  Y = X @ W^T + b -----------------------------
// TRANS=0: Y[m][n] (row-major [B*S][E]);  TRANS=1: Y[b][n][s] ([B][E][S])
template <int TRANS>
__global__ void __launch_bounds__(128)
qkv_gemm(const float* __restrict__ X, const float* __restrict__ W,
         const float* __restrict__ bias, __bf16* __restrict__ Y) {
  __shared__ alignas(16) __bf16 As[64][40];
  __shared__ alignas(16) __bf16 Bs[64][40];
  const int tid = threadIdx.x, lane = tid & 31, wv = tid >> 5;
  const int m0 = blockIdx.x * 64, n0 = blockIdx.y * 64;

  v8f acc[4] = {};
  for (int kc = 0; kc < EMBED; kc += 32) {
    {
      const int row = tid >> 1, koff = (tid & 1) * 16;
      const float4* xp = (const float4*)(X + (size_t)(m0 + row) * EMBED + kc + koff);
      const float4* wp = (const float4*)(W + (size_t)(n0 + row) * EMBED + kc + koff);
      float fx[16], fw[16];
      *(float4*)&fx[0] = xp[0]; *(float4*)&fx[4] = xp[1];
      *(float4*)&fx[8] = xp[2]; *(float4*)&fx[12] = xp[3];
      *(float4*)&fw[0] = wp[0]; *(float4*)&fw[4] = wp[1];
      *(float4*)&fw[8] = wp[2]; *(float4*)&fw[12] = wp[3];
      v8bf xlo, xhi, wlo, whi;
#pragma unroll
      for (int i = 0; i < 8; ++i) {
        xlo[i] = (__bf16)fx[i]; xhi[i] = (__bf16)fx[8 + i];
        wlo[i] = (__bf16)fw[i]; whi[i] = (__bf16)fw[8 + i];
      }
      *(v8bf*)&As[row][koff] = xlo; *(v8bf*)&As[row][koff + 8] = xhi;
      *(v8bf*)&Bs[row][koff] = wlo; *(v8bf*)&Bs[row][koff + 8] = whi;
    }
    __syncthreads();
    const v16bf a = load_a_frag(&As[wv * 16][0], 40, lane);
#pragma unroll
    for (int nt = 0; nt < 4; ++nt) {
      const v16bf b = load_b_frag(&Bs[nt * 16][0], 40, lane);
      acc[nt] = __builtin_amdgcn_wmma_f32_16x16x32_bf16(
          false, a, false, b, (short)0, acc[nt], false, false);
    }
    __syncthreads();
  }
  const int nl = lane & 15, half = lane >> 4;
#pragma unroll
  for (int nt = 0; nt < 4; ++nt) {
    const int n = n0 + nt * 16 + nl;
    const float bv = bias[n];
#pragma unroll
    for (int r = 0; r < 8; ++r) {
      const int m = m0 + wv * 16 + r + 8 * half;
      const __bf16 yv = (__bf16)(acc[nt][r] + bv);
      if (TRANS == 0) {
        Y[(size_t)m * EMBED + n] = yv;
      } else {
        const int bb = m >> 12, s = m & (SEQ - 1);
        Y[((size_t)bb * EMBED + n) * SEQ + s] = yv;
      }
    }
  }
}

// ---- Pass 1: softmax row stats (max, sumexp) ------------------------------
__global__ void __launch_bounds__(128)
attn_stats(const __bf16* __restrict__ Qb, const __bf16* __restrict__ Kb,
           float* __restrict__ Mrow, float* __restrict__ Lrow) {
  __shared__ alignas(16) __bf16 Qc[2][64][72];
  __shared__ alignas(16) __bf16 Kc[2][64][72];
  const int tid = threadIdx.x, lane = tid & 31, wv = tid >> 5;
  const int bb = blockIdx.y, q0 = blockIdx.x * 64;
  const __bf16* Qp = Qb + ((size_t)bb * SEQ + q0) * EMBED;
  const __bf16* Kp = Kb + (size_t)bb * SEQ * EMBED;

  float mrun[8], lrun[8];
#pragma unroll
  for (int r = 0; r < 8; ++r) { mrun[r] = -1e30f; lrun[r] = 0.f; }

  for (int j = 0; j < SEQ; j += 64) {
    const __bf16* Kj = Kp + (size_t)j * EMBED;
    v8f sacc[4] = {};
    fill_qk(Qp, Kj, tid, Qc[0], Kc[0], 0);
    for (int c = 0; c < NCHUNK; ++c) {
      const int cur = c & 1;
      if (c + 1 < NCHUNK) {
        fill_qk(Qp, Kj, tid, Qc[cur ^ 1], Kc[cur ^ 1], (c + 1) * KCHUNK);
        async_wait<8>();
      } else {
        async_wait<0>();
      }
      __syncthreads();
#pragma unroll
      for (int ks = 0; ks < KCHUNK; ks += 32) {
        const v16bf a = load_a_frag(&Qc[cur][wv * 16][ks], 72, lane);
#pragma unroll
        for (int t = 0; t < 4; ++t) {
          const v16bf b = load_b_frag(&Kc[cur][t * 16][ks], 72, lane);
          sacc[t] = __builtin_amdgcn_wmma_f32_16x16x32_bf16(
              false, a, false, b, (short)0, sacc[t], false, false);
        }
      }
      __syncthreads();
    }
#pragma unroll
    for (int r = 0; r < 8; ++r) {
      float mx = -1e30f;
#pragma unroll
      for (int t = 0; t < 4; ++t) mx = fmaxf(mx, sacc[t][r]);
#pragma unroll
      for (int off = 1; off < 16; off <<= 1) mx = fmaxf(mx, __shfl_xor(mx, off, 32));
      mx *= SCALE;
      const float mnew = fmaxf(mrun[r], mx);
      float ssum = 0.f;
#pragma unroll
      for (int t = 0; t < 4; ++t) ssum += __expf(sacc[t][r] * SCALE - mnew);
#pragma unroll
      for (int off = 1; off < 16; off <<= 1) ssum += __shfl_xor(ssum, off, 32);
      lrun[r] = lrun[r] * __expf(mrun[r] - mnew) + ssum;
      mrun[r] = mnew;
    }
  }
  const int nl = lane & 15, half = lane >> 4;
  if (nl == 0) {
#pragma unroll
    for (int r = 0; r < 8; ++r) {
      const int row = q0 + wv * 16 + r + 8 * half;
      Mrow[(size_t)bb * SEQ + row] = mrun[r];
      Lrow[(size_t)bb * SEQ + row] = lrun[r];
    }
  }
}

// ---- Pass 2: O = softmax(QK^T) V for a 256-wide embed slice ---------------
__global__ void __launch_bounds__(128)
attn_out(const __bf16* __restrict__ Qb, const __bf16* __restrict__ Kb,
         const __bf16* __restrict__ Vtb, const float* __restrict__ Mrow,
         const float* __restrict__ Lrow, float* __restrict__ O) {
  __shared__ alignas(16) __bf16 Qc[2][64][72];
  __shared__ alignas(16) __bf16 Kc[2][64][72];
  __shared__ alignas(16) __bf16 Pc[64][72];      // probs, row-major [q][key]
  __shared__ alignas(16) __bf16 Vc[ETILE][72];   // V^T slice, [e][key]
  const int tid = threadIdx.x, lane = tid & 31, wv = tid >> 5;
  const int bb = blockIdx.z, q0 = blockIdx.x * 64, e0 = blockIdx.y * ETILE;
  const int nl = lane & 15, half = lane >> 4;
  const __bf16* Qp = Qb + ((size_t)bb * SEQ + q0) * EMBED;
  const __bf16* Kp = Kb + (size_t)bb * SEQ * EMBED;
  const __bf16* Vt = Vtb + (size_t)bb * EMBED * SEQ;

  float mst[8], lst[8];
#pragma unroll
  for (int r = 0; r < 8; ++r) {
    const int row = q0 + wv * 16 + r + 8 * half;
    mst[r] = Mrow[(size_t)bb * SEQ + row];
    lst[r] = 1.0f / Lrow[(size_t)bb * SEQ + row];
  }

  v8f oacc[16] = {};
  for (int j = 0; j < SEQ; j += 64) {
    const __bf16* Kj = Kp + (size_t)j * EMBED;
    v8f sacc[4] = {};
    fill_qk(Qp, Kj, tid, Qc[0], Kc[0], 0);
    for (int c = 0; c < NCHUNK; ++c) {
      const int cur = c & 1;
      if (c + 1 < NCHUNK) {
        fill_qk(Qp, Kj, tid, Qc[cur ^ 1], Kc[cur ^ 1], (c + 1) * KCHUNK);
        async_wait<8>();
      } else {
        async_wait<0>();
      }
      __syncthreads();
#pragma unroll
      for (int ks = 0; ks < KCHUNK; ks += 32) {
        const v16bf a = load_a_frag(&Qc[cur][wv * 16][ks], 72, lane);
#pragma unroll
        for (int t = 0; t < 4; ++t) {
          const v16bf b = load_b_frag(&Kc[cur][t * 16][ks], 72, lane);
          sacc[t] = __builtin_amdgcn_wmma_f32_16x16x32_bf16(
              false, a, false, b, (short)0, sacc[t], false, false);
        }
      }
      __syncthreads();
    }
    // probabilities -> LDS (own-wave rows only)
#pragma unroll
    for (int t = 0; t < 4; ++t)
#pragma unroll
      for (int r = 0; r < 8; ++r) {
        const float p = __expf(sacc[t][r] * SCALE - mst[r]) * lst[r];
        Pc[wv * 16 + r + 8 * half][t * 16 + nl] = (__bf16)p;
      }
    // V^T slice (ETILE e-rows x 64 keys) -> LDS, coalesced 16B chunks
    {
      const int er = tid * 2;
      const __bf16* vs = Vt + (size_t)(e0 + er) * SEQ + j;
#pragma unroll
      for (int ck = 0; ck < 8; ++ck) {
        copy16(vs + ck * 8,       &Vc[er][ck * 8]);
        copy16(vs + SEQ + ck * 8, &Vc[er + 1][ck * 8]);
      }
    }
    async_wait<0>();
    __syncthreads();
    // P @ V
#pragma unroll
    for (int kc = 0; kc < 2; ++kc) {
      const v16bf a = load_a_frag(&Pc[wv * 16][kc * 32], 72, lane);
#pragma unroll
      for (int nt = 0; nt < 16; ++nt) {
        const v16bf b = load_b_frag(&Vc[nt * 16][kc * 32], 72, lane);
        oacc[nt] = __builtin_amdgcn_wmma_f32_16x16x32_bf16(
            false, a, false, b, (short)0, oacc[nt], false, false);
      }
    }
    __syncthreads();
  }
#pragma unroll
  for (int nt = 0; nt < 16; ++nt)
#pragma unroll
    for (int r = 0; r < 8; ++r) {
      const int m = q0 + wv * 16 + r + 8 * half;
      O[((size_t)bb * SEQ + m) * EMBED + e0 + nt * 16 + nl] = oacc[nt][r];
    }
}

// ---------------------------------------------------------------------------
extern "C" void kernel_launch(void* const* d_in, const int* in_sizes, int n_in,
                              void* d_out, int out_size, void* d_ws, size_t ws_size,
                              hipStream_t stream) {
  (void)in_sizes; (void)n_in; (void)out_size; (void)ws_size;
  const float* query = (const float*)d_in[0];
  const float* key   = (const float*)d_in[1];
  const float* value = (const float*)d_in[2];
  const float* Wq = (const float*)d_in[3];
  const float* bq = (const float*)d_in[4];
  const float* Wk = (const float*)d_in[5];
  const float* bk = (const float*)d_in[6];
  const float* Wv = (const float*)d_in[7];
  const float* bv = (const float*)d_in[8];
  float* Out = (float*)d_out;

  char* ws = (char*)d_ws;
  __bf16* Qb  = (__bf16*)(ws);
  __bf16* Kb  = (__bf16*)(ws + ((size_t)32 << 20));
  __bf16* Vtb = (__bf16*)(ws + ((size_t)64 << 20));   // transposed [B][E][S]
  float*  Mrow = (float*)(ws + ((size_t)96 << 20));
  float*  Lrow = Mrow + (size_t)NBATCH * SEQ;

  const dim3 blk(128);
  qkv_gemm<0><<<dim3(256, 16), blk, 0, stream>>>(query, Wq, bq, Qb);
  qkv_gemm<0><<<dim3(256, 16), blk, 0, stream>>>(key,   Wk, bk, Kb);
  qkv_gemm<1><<<dim3(256, 16), blk, 0, stream>>>(value, Wv, bv, Vtb);
  attn_stats<<<dim3(SEQ / 64, NBATCH), blk, 0, stream>>>(Qb, Kb, Mrow, Lrow);
  attn_out<<<dim3(SEQ / 64, EMBED / ETILE, NBATCH), blk, 0, stream>>>(
      Qb, Kb, Vtb, Mrow, Lrow, Out);
}
